// AdditiveAttention_48859547959476
// MI455X (gfx1250) — matrix operand order repaired and checked by reading, not compiled, not run on hardware
//
#include <hip/hip_runtime.h>

typedef __attribute__((ext_vector_type(16))) _Float16 v16h;
typedef __attribute__((ext_vector_type(8)))  float    v8f;

#define DHID 64

// ---------------- CDNA5 async global->LDS support (guarded) ----------------
#if defined(__gfx1250__) && __has_builtin(__builtin_amdgcn_global_load_async_to_lds_b128)
#define HAVE_ASYNC_LDS 1
typedef int vi4 __attribute__((vector_size(16)));   // matches builtin's V4i
#define AS1_ __attribute__((address_space(1)))
#define AS3_ __attribute__((address_space(3)))
// builtin signature: (v4i as1*, v4i as3*, imm offset, imm cpol)
#define ASYNC_COPY_B128(gp, lp) \
    __builtin_amdgcn_global_load_async_to_lds_b128( \
        (AS1_ vi4*)(vi4*)(gp), (AS3_ vi4*)(vi4*)(lp), 0, 0)
#define ASYNC_COPY_B32(gp, lp) \
    __builtin_amdgcn_global_load_async_to_lds_b32( \
        (AS1_ int*)(int*)(gp), (AS3_ int*)(int*)(lp), 0, 0)
#if __has_builtin(__builtin_amdgcn_s_wait_asynccnt)
#define WAIT_ASYNC(n) __builtin_amdgcn_s_wait_asynccnt(n)
#else
#define WAIT_ASYNC(n) asm volatile("s_wait_asynccnt %0" ::"n"(n) : "memory")
#endif
#else
#define HAVE_ASYNC_LDS 0
#endif

// ---------------------------------------------------------------------------
// Kernel 1: projection GEMM  Y[M x 64] = X[M x Kdim] @ W[Kdim x 64]
// One wave per 16x16 output tile, 4 waves per block. f16 WMMA, f32 accum.
// ---------------------------------------------------------------------------
__global__ __launch_bounds__(128) void proj_gemm_kernel(
    const float* __restrict__ X, const float* __restrict__ W,
    float* __restrict__ Y, int M, int Kdim, int Ndim)
{
    const int lane  = threadIdx.x & 31;
    const int wave  = threadIdx.x >> 5;
    const int mtile = blockIdx.x * 4 + wave;   // 16-row tile index
    const int ntile = blockIdx.y;              // 16-col tile index
    if (mtile * 16 >= M) return;               // uniform per wave -> EXEC stays full

    const int half = lane >> 4;                // 0: lanes 0-15, 1: lanes 16-31
    const int l15  = lane & 15;
    const int m    = mtile * 16 + l15;
    const int n    = ntile * 16 + l15;

    v8f acc = {};
    for (int k0 = 0; k0 < Kdim; k0 += 32) {
        union { v16h v; _Float16 h[16]; } a, b;
        // A fragment: row m, K = k0 + {half*8+e , half*8+16+e}
        const float* xr = X + (size_t)m * Kdim + k0 + half * 8;
        #pragma unroll
        for (int e = 0; e < 8; ++e) {
            a.h[e]     = (_Float16)xr[e];
            a.h[8 + e] = (_Float16)xr[16 + e];
        }
        // B fragment: col n, K = k0 + {half*8+e , half*8+16+e}  (column of W, stride Ndim)
        const float* wc = W + (size_t)(k0 + 8 * half) * Ndim + n;
        #pragma unroll
        for (int e = 0; e < 16; ++e)
            b.h[e] = (_Float16)wc[(size_t)(e < 8 ? e : e + 8) * Ndim];

        acc = __builtin_amdgcn_wmma_f32_16x16x32_f16(
                  false, a.v, false, b.v, (short)0, acc, false, false);
    }
    // C layout: VGPR r -> row r + 8*half, col lane&15
    #pragma unroll
    for (int r = 0; r < 8; ++r) {
        const int mm = mtile * 16 + r + 8 * half;
        Y[(size_t)mm * Ndim + n] = acc[r];
    }
}

// ---------------------------------------------------------------------------
// Kernel 2: additive-attention scores + length mask.
// scores[b,q,k] = sum_h wv[h]*tanh(qp[b,q,h]+kp[b,k,h]); masked -> -1e9.
// Block = (16 q-rows) x (128 k-cols), 256 threads; q-row in registers,
// k-tile staged in LDS via async b128 copies; tanh via v_exp_f32 + v_rcp_f32.
// ---------------------------------------------------------------------------
__global__ __launch_bounds__(256) void scores_kernel(
    const float* __restrict__ qp, const float* __restrict__ kp,
    const float* __restrict__ wv, const int* __restrict__ valid_lens,
    float* __restrict__ scores, int NQ, int NK)
{
    const int b  = blockIdx.z;
    const int qt = blockIdx.y;     // 16 q rows per block
    const int kt = blockIdx.x;     // 128 k cols per block

    __shared__ __align__(16) float s_k[128][DHID];   // 32 KB (of 320 KB/WGP)
    __shared__ float s_wv[DHID];

    const int tid = threadIdx.x;
    if (tid < DHID) s_wv[tid] = wv[tid];

    // stage 128x64 k-projection tile
    const float* ktile = kp + ((size_t)b * NK + (size_t)kt * 128) * DHID;
#if HAVE_ASYNC_LDS
    #pragma unroll
    for (int i = 0; i < 8; ++i) {
        const int idx = tid + 256 * i;               // 0..2047 b128 chunks
        ASYNC_COPY_B128((const char*)ktile + idx * 16,
                        (char*)&s_k[0][0] + idx * 16);
    }
    WAIT_ASYNC(0);
#else
    {
        const float4* kpv = (const float4*)ktile;
        float4* skv = (float4*)&s_k[0][0];
        #pragma unroll
        for (int i = 0; i < 8; ++i) skv[tid + 256 * i] = kpv[tid + 256 * i];
    }
#endif

    // q-row for this thread stays in registers
    const int q = tid & 15;
    float qrow[DHID];
    const float4* qr4 = (const float4*)(qp + ((size_t)b * NQ + qt * 16 + q) * DHID);
    #pragma unroll
    for (int h4 = 0; h4 < DHID / 4; ++h4) {
        float4 t = qr4[h4];
        qrow[4 * h4 + 0] = t.x; qrow[4 * h4 + 1] = t.y;
        qrow[4 * h4 + 2] = t.z; qrow[4 * h4 + 3] = t.w;
    }
    __syncthreads();

    const int   vlen  = valid_lens[b];
    const float L2E2  = 2.8853900817779268f;   // 2*log2(e)
    float* srow = scores + ((size_t)b * NQ + qt * 16 + q) * NK + (size_t)kt * 128;

    #pragma unroll 1
    for (int kk = 0; kk < 8; ++kk) {
        const int kl = (tid >> 4) + kk * 16;         // 0..127, unique per (q,kl)
        const float4* k4 = (const float4*)s_k[kl];   // broadcast within 16-thread group
        const float4* w4 = (const float4*)s_wv;
        float s = 0.f;
        #pragma unroll
        for (int h4 = 0; h4 < DHID / 4; ++h4) {
            float4 kv = k4[h4];
            float4 wvv = w4[h4];
            float xs[4] = { qrow[4*h4+0] + kv.x, qrow[4*h4+1] + kv.y,
                            qrow[4*h4+2] + kv.z, qrow[4*h4+3] + kv.w };
            float ws[4] = { wvv.x, wvv.y, wvv.z, wvv.w };
            #pragma unroll
            for (int j = 0; j < 4; ++j) {
                float e = __builtin_amdgcn_exp2f(xs[j] * L2E2);        // v_exp_f32
                float t = 1.f - 2.f * __builtin_amdgcn_rcpf(e + 1.f);  // v_rcp_f32
                s = __builtin_fmaf(ws[j], t, s);
            }
        }
        const int kg = kt * 128 + kl;
        srow[kl] = (kg < vlen) ? s : -1e9f;
    }
}

// ---------------------------------------------------------------------------
// Kernel 3: masked softmax over NK=2048 per (b,q) row -> f16 probabilities.
// One 256-thread block per row; wave32 shuffle reduce + cross-wave LDS.
// ---------------------------------------------------------------------------
__global__ __launch_bounds__(256) void softmax_kernel(
    const float* __restrict__ scores, _Float16* __restrict__ attn, int NK)
{
    const int row = blockIdx.x;                  // b*NQ + q
    const float* s = scores + (size_t)row * NK;
    _Float16*    a = attn   + (size_t)row * NK;
    const int tid = threadIdx.x;

    __shared__ float red[8];

    float v[8];
    float lmax = -3.4e38f;
    #pragma unroll
    for (int i = 0; i < 8; ++i) { v[i] = s[tid + 256 * i]; lmax = fmaxf(lmax, v[i]); }
    #pragma unroll
    for (int off = 16; off > 0; off >>= 1)
        lmax = fmaxf(lmax, __shfl_xor(lmax, off, 32));
    if ((tid & 31) == 0) red[tid >> 5] = lmax;
    __syncthreads();
    float m = red[0];
    #pragma unroll
    for (int w = 1; w < 8; ++w) m = fmaxf(m, red[w]);

    const float L2E = 1.4426950408889634f;
    float lsum = 0.f;
    #pragma unroll
    for (int i = 0; i < 8; ++i) {
        v[i] = __builtin_amdgcn_exp2f((v[i] - m) * L2E);
        lsum += v[i];
    }
    #pragma unroll
    for (int off = 16; off > 0; off >>= 1) lsum += __shfl_xor(lsum, off, 32);
    __syncthreads();
    if ((tid & 31) == 0) red[tid >> 5] = lsum;
    __syncthreads();
    float total = 0.f;
    #pragma unroll
    for (int w = 0; w < 8; ++w) total += red[w];
    const float inv = __builtin_amdgcn_rcpf(total);
    #pragma unroll
    for (int i = 0; i < 8; ++i) a[tid + 256 * i] = (_Float16)(v[i] * inv);
}

// ---------------------------------------------------------------------------
// Kernel 4: O[b] = attn[b] (NQ x NK, f16) @ V[b] (NK x DV, f32->f16).
// Block = one 16-row M tile, 8 waves = 8 N tiles (DV=128). A and V tiles
// double-buffered in LDS via async global->LDS copies (5 async instrs per
// wave per tile: 4x b128 for V, 1x b32 for A); s_wait_asynccnt 5 keeps the
// next tile's copies in flight while the current WMMA executes.
// ---------------------------------------------------------------------------
#define VPADF 132  // padded f32 row stride (528 B, 16B-aligned; dodges conflicts)

__global__ __launch_bounds__(256) void av_gemm_kernel(
    const _Float16* __restrict__ attn, const float* __restrict__ V,
    float* __restrict__ out, int NQ, int NK, int DV)
{
    const int b     = blockIdx.y;
    const int mtile = blockIdx.x;
    const int tid   = threadIdx.x;
    const int lane  = tid & 31;
    const int wave  = tid >> 5;        // N tile 0..7
    const int half  = lane >> 4;
    const int l15   = lane & 15;

    __shared__ __align__(16) float    s_vf[2][32 * VPADF];  // ~33 KB
    __shared__ __align__(16) _Float16 s_a [2][16 * 32];     //   2 KB

    const _Float16* A  = attn + ((size_t)b * NQ + mtile * 16) * NK;
    const float*    Vb = V + (size_t)b * NK * DV;
    const int       n  = wave * 16 + l15;
    const int       T  = NK / 32;

    // per-thread staging coordinates
    const int vr = tid >> 3;            // V row 0..31 (8 threads per row, 4 b128 chunks each)
    const int vc = (tid & 7) * 16;      // V col base (4 chunks of 4 floats)
    const int ar = tid >> 4;            // A row 0..15
    const int ac = (tid & 15) * 2;      // A col (2 halves per b32)

#if HAVE_ASYNC_LDS
    auto issue_tile = [&](int t, int bufsel) {
        const float* vsrc = Vb + ((size_t)t * 32 + vr) * DV + vc;
        float*       vdst = &s_vf[bufsel][vr * VPADF + vc];
        #pragma unroll
        for (int c = 0; c < 4; ++c)
            ASYNC_COPY_B128(vsrc + c * 4, vdst + c * 4);
        ASYNC_COPY_B32(A + (size_t)ar * NK + t * 32 + ac,
                       &s_a[bufsel][ar * 32 + ac]);
    };
#else
    auto issue_tile = [&](int t, int bufsel) {
        const float* vsrc = Vb + ((size_t)t * 32 + vr) * DV + vc;
        float*       vdst = &s_vf[bufsel][vr * VPADF + vc];
        #pragma unroll
        for (int c = 0; c < 4; ++c)
            *(float4*)(vdst + c * 4) = *(const float4*)(vsrc + c * 4);
        *(unsigned*)&s_a[bufsel][ar * 32 + ac] =
            *(const unsigned*)(A + (size_t)ar * NK + t * 32 + ac);
    };
#endif

    v8f acc = {};
    int buf = 0;
    issue_tile(0, 0);
    for (int t = 0; t < T; ++t) {
        if (t + 1 < T) {
            issue_tile(t + 1, buf ^ 1);
#if HAVE_ASYNC_LDS
            WAIT_ASYNC(5);      // tile t landed; tile t+1 still in flight
#endif
        } else {
#if HAVE_ASYNC_LDS
            WAIT_ASYNC(0);
#endif
        }
        __syncthreads();

        union { v16h v; _Float16 h[16]; } a, bm;
        #pragma unroll
        for (int e = 0; e < 8; ++e) {
            a.h[e]     = s_a[buf][l15 * 32 + half * 8 + e];
            a.h[8 + e] = s_a[buf][l15 * 32 + half * 8 + 16 + e];
        }
        #pragma unroll
        for (int e = 0; e < 16; ++e)
            bm.h[e] = (_Float16)s_vf[buf][((e < 8 ? e : e + 8) + 8 * half) * VPADF + n];

        acc = __builtin_amdgcn_wmma_f32_16x16x32_f16(
                  false, a.v, false, bm.v, (short)0, acc, false, false);

        __syncthreads();        // everyone done reading buf before it is refilled
        buf ^= 1;
    }

    #pragma unroll
    for (int r = 0; r < 8; ++r) {
        const int m = mtile * 16 + r + 8 * half;
        out[((size_t)b * NQ + m) * DV + n] = acc[r];
    }
}

// ---------------------------------------------------------------------------
extern "C" void kernel_launch(void* const* d_in, const int* in_sizes, int n_in,
                              void* d_out, int out_size, void* d_ws, size_t ws_size,
                              hipStream_t stream)
{
    const int B = 16, NQ = 128, NK = 2048, DQ = 256, DK = 256, DV = 128;

    const float* queries    = (const float*)d_in[0];
    const float* keys       = (const float*)d_in[1];
    const float* values     = (const float*)d_in[2];
    const int*   valid_lens = (const int*)d_in[3];
    const float* Wq         = (const float*)d_in[4];
    const float* Wk         = (const float*)d_in[5];
    const float* wv         = (const float*)d_in[6];
    float*       out        = (float*)d_out;

    // workspace: qp | kp | scores(f32) | attn(f16)  == 32 MB total
    float*     qp     = (float*)d_ws;
    float*     kp     = qp + (size_t)B * NQ * DHID;
    float*     scores = kp + (size_t)B * NK * DHID;
    _Float16*  attn   = (_Float16*)(scores + (size_t)B * NQ * NK);

    // 1) projections (WMMA f16 GEMMs)
    dim3 gq((B * NQ) / 64, DHID / 16);
    proj_gemm_kernel<<<gq, 128, 0, stream>>>(queries, Wq, qp, B * NQ, DQ, DHID);
    dim3 gk((B * NK) / 64, DHID / 16);
    proj_gemm_kernel<<<gk, 128, 0, stream>>>(keys, Wk, kp, B * NK, DK, DHID);

    // 2) tanh scores + mask (dominant cost: 268M transcendental evals)
    dim3 gs(NK / 128, NQ / 16, B);
    scores_kernel<<<gs, 256, 0, stream>>>(qp, kp, wv, valid_lens, scores, NQ, NK);

    // 3) masked softmax -> f16 probabilities
    softmax_kernel<<<B * NQ, 256, 0, stream>>>(scores, attn, NK);

    // 4) attn @ V (WMMA f16 GEMM, async double-buffered LDS staging)
    dim3 go(NQ / 16, B);
    av_gemm_kernel<<<go, 256, 0, stream>>>(attn, values, out, NQ, NK, DV);
}
